// T5Attention_15590731284968
// MI455X (gfx1250) — matrix operand order repaired and checked by reading, not compiled, not run on hardware
//
#include <hip/hip_runtime.h>
#include <hip/hip_bf16.h>
#include <cstdint>

// ---------------------------------------------------------------------------
// T5 attention for MI455X (gfx1250, wave32)
//   B=4, S=2048, H=16, D=64, DM=1024
//   - QKV projection: WMMA f32_16x16x32_f16, fp32->fp16 staged through LDS.
//     V is written TRANSPOSED ([B,H,D,S]) so attention needs no transposes.
//   - Attention: flash-style, K/V tiles streamed global->LDS by the Tensor
//     Data Mover (tensor_load_to_lds, TENSORcnt), double buffered.
// ---------------------------------------------------------------------------

typedef _Float16 f16;
typedef __attribute__((ext_vector_type(16))) _Float16 v16h;
typedef __attribute__((ext_vector_type(8)))  float    v8f;
typedef __attribute__((ext_vector_type(4)))  uint32_t v4u;
typedef __attribute__((ext_vector_type(8)))  uint32_t v8u;

#define T5_B  4
#define T5_S  2048
#define T5_H  16
#define T5_D  64
#define T5_DM 1024

// ------------------------- WMMA helpers ------------------------------------

__device__ __forceinline__ v8f wmma16(v16h a, v16h b, v8f c) {
  return __builtin_amdgcn_wmma_f32_16x16x32_f16(
      false, a, false, b, (short)0, c, false, false);
}

// A-fragment (16x32 f16, MxK) per ISA 7.12.2; src row-major [M][ld].
__device__ __forceinline__ v16h load_a_frag(const f16* p, int ld, int m0, int k0) {
  const int lane = threadIdx.x & 31;
  const int hi   = (lane >> 4) << 3; // 0 or 8
  const f16* row = p + (size_t)(m0 + (lane & 15)) * ld + k0;
  v16h a;
#pragma unroll
  for (int v = 0; v < 8; ++v) {
    int kk = ((v < 4) ? (2 * v) : (16 + 2 * (v - 4))) + hi;
    a[2 * v]     = row[kk];
    a[2 * v + 1] = row[kk + 1];
  }
  return a;
}

// B-fragment (32x16 f16, KxN), B[k][n] = src[n][k], src row-major [N][ld]:
// lane n (0-15): K=0..7,16..23 ; lane n+16: K=8..15,24..31 (A-frag k order).
__device__ __forceinline__ v16h load_b_frag(const f16* p, int ld, int n0, int k0) {
  const int lane = threadIdx.x & 31;
  const f16* row = p + (size_t)(n0 + (lane & 15)) * ld + k0 + ((lane >> 4) << 3);
  v16h b;
#pragma unroll
  for (int i = 0; i < 16; ++i) b[i] = row[(i < 8) ? i : (i + 8)];
  return b;
}

// ------------------------- Tensor Data Mover -------------------------------
// Build a D# for a 2-D f16 tile copy (global -> LDS) and issue
// tensor_load_to_lds (VADDR2/3 omitted -> 2-D form). Per CDNA5 ISA:
//   group0: [1:0]=count, [63:32]=lds_addr, [120:64]=global_addr, [127:126]=2
//   group1: data_size(2B), tensor_dim0/1, tile_dim0/1, tensor_dim0_stride
// All operands are workgroup-uniform -> SGPRs.

__device__ __forceinline__ void tdm_load_2d(const void* gptr, void* lptr,
                                            uint32_t tensor_d0, uint32_t tensor_d1,
                                            uint32_t tile_d0, uint32_t tile_d1,
                                            uint64_t stride0_elems) {
  uint64_t ga = (uint64_t)(uintptr_t)gptr;
  v4u g0;
  g0[0] = 1u;                                  // count = 1 (valid, user mode)
  g0[1] = (uint32_t)(uintptr_t)lptr;           // LDS byte offset (flat low 32)
  g0[2] = (uint32_t)ga;                        // global_addr[31:0]
  g0[3] = ((uint32_t)(ga >> 32) & 0x01FFFFFFu) // global_addr[56:32]
          | (2u << 30);                        // type = 2 ("image")
  v8u g1;
  g1[0] = (1u << 16);                          // data_size = 1 -> 2 bytes
  g1[1] = (tensor_d0 & 0xFFFFu) << 16;         // tensor_dim0[15:0] @ bits63:48
  g1[2] = (tensor_d0 >> 16) | ((tensor_d1 & 0xFFFFu) << 16);
  g1[3] = (tensor_d1 >> 16) | (tile_d0 << 16); // tile_dim0 @ bits127:112
  g1[4] = tile_d1;                             // tile_dim1 ; tile_dim2 = 0
  g1[5] = (uint32_t)(stride0_elems & 0xFFFFFFFFu);
  g1[6] = (uint32_t)((stride0_elems >> 32) & 0xFFFFu);
  g1[7] = 0u;                                  // tensor_dim1_stride unused (2D)
  asm volatile("tensor_load_to_lds %0, %1" :: "s"(g0), "s"(g1) : "memory");
}

// ------------------------- kernel 1: fused QKV projection -------------------
// C[m,n] = sum_k hidden[m,k] * W[n,k]. Q,K out fp16 [B,H,S,D]; V out [B,H,D,S].

__global__ __launch_bounds__(256) void t5_proj_kernel(
    const float* __restrict__ hidden,
    const float* __restrict__ Wq, const float* __restrict__ Wk,
    const float* __restrict__ Wv,
    f16* __restrict__ qout, f16* __restrict__ kout, f16* __restrict__ vout) {
  constexpr int BM = 128, BN = 64, BK = 32;

  const float* W;
  f16* dst;
  if (blockIdx.z == 0)      { W = Wq; dst = qout; }
  else if (blockIdx.z == 1) { W = Wk; dst = kout; }
  else                      { W = Wv; dst = vout; }
  const bool vtrans = (blockIdx.z == 2);

  __shared__ f16 As[BM * BK];  // 8 KB
  __shared__ f16 Bs[BN * BK];  // 4 KB

  const int tid  = threadIdx.x;
  const int lane = tid & 31;
  const int w    = tid >> 5;
  const int wm   = w & 3;
  const int wn   = w >> 2;
  const int m0b  = blockIdx.y * BM;
  const int n0b  = blockIdx.x * BN;

  v8f acc[2][2] = {};

  for (int k0 = 0; k0 < T5_DM; k0 += BK) {
    {
      int row = tid >> 1;
      int col = (tid & 1) * 16;
      const float* src = hidden + (size_t)(m0b + row) * T5_DM + k0 + col;
      f16* d = As + row * BK + col;
#pragma unroll
      for (int i = 0; i < 16; ++i) d[i] = (f16)src[i];
    }
    {
      int row = tid >> 2;
      int col = (tid & 3) * 8;
      const float* src = W + (size_t)(n0b + row) * T5_DM + k0 + col;
      f16* d = Bs + row * BK + col;
#pragma unroll
      for (int i = 0; i < 8; ++i) d[i] = (f16)src[i];
    }
    __syncthreads();

    v16h a0 = load_a_frag(As, BK, wm * 32 + 0, 0);
    v16h a1 = load_a_frag(As, BK, wm * 32 + 16, 0);
    v16h b0 = load_b_frag(Bs, BK, wn * 32 + 0, 0);
    v16h b1 = load_b_frag(Bs, BK, wn * 32 + 16, 0);
    acc[0][0] = wmma16(a0, b0, acc[0][0]);
    acc[0][1] = wmma16(a0, b1, acc[0][1]);
    acc[1][0] = wmma16(a1, b0, acc[1][0]);
    acc[1][1] = wmma16(a1, b1, acc[1][1]);
    __syncthreads();
  }

  // Epilogue. A 128-row block never straddles a batch boundary (128 | 2048),
  // so batch, head base and the two layout strides are block-constant:
  //   idx = headbase + s*ss + d*sd,  (ss,sd) = vtrans ? (1,S) : (D,1).
  const int hcol  = n0b >> 6;          // head index (BN == D == 64)
  const int ln    = lane & 15;
  const int lhi   = (lane >> 4) << 3;
  const int bidx  = m0b >> 11;         // m / S
  const int sbase = m0b & (T5_S - 1);  // m % S for r == 0 of wave 0
  const size_t headbase =
      ((size_t)bidx * T5_H + hcol) * (size_t)(T5_S * T5_D);
  const size_t ss = vtrans ? (size_t)1 : (size_t)T5_D;
  const size_t sd = vtrans ? (size_t)T5_S : (size_t)1;

#pragma unroll
  for (int ti = 0; ti < 2; ++ti)
#pragma unroll
    for (int tj = 0; tj < 2; ++tj) {
      int d = wn * 32 + tj * 16 + ln;  // dim within head
      const size_t base_d = headbase + (size_t)d * sd;
#pragma unroll
      for (int r = 0; r < 8; ++r) {
        int s = sbase + wm * 32 + ti * 16 + r + lhi;
        dst[base_d + (size_t)s * ss] = (f16)acc[ti][tj][r];
      }
    }
}

// ------------------------- kernel 2: flash attention ------------------------
// Block = (b, h, 64 q rows); 4 waves x 16 rows with private online softmax.
// K/V 64-wide tiles double-buffered in LDS via the Tensor Data Mover.

__global__ __launch_bounds__(128) void t5_attn_kernel(
    const f16* __restrict__ qg, const f16* __restrict__ kg,
    const f16* __restrict__ vtg, const float* __restrict__ pb,
    float* __restrict__ out) {
  const int bh   = blockIdx.y;
  const int b    = bh >> 4;
  const int h    = bh & (T5_H - 1);
  const int tid  = threadIdx.x;
  const int lane = tid & 31;
  const int w    = tid >> 5;  // wave 0..3
  const int ln   = lane & 15;
  const int lhi  = (lane >> 4) << 3;

  const size_t headoff = ((size_t)b * T5_H + h) * T5_S * T5_D;
  const f16* qh  = qg  + headoff;   // [S][D]
  const f16* kh  = kg  + headoff;   // [S][D]
  const f16* vth = vtg + headoff;   // [D][S] (transposed)
  const float* pbh = pb + (size_t)h * T5_S * T5_S;

  const int qs0 = blockIdx.x * 64 + w * 16;

  __shared__ f16 Ks[2][64 * 64];    // K tile [t][d], 2 x 8 KB
  __shared__ f16 Vs[2][64 * 64];    // V tile [d][t], 2 x 8 KB
  __shared__ f16 Pbuf[4][16 * 64];  // per-wave P tile, 8 KB

  // Q fragments stay in registers the whole kernel (D=64 -> 2 K-steps).
  v16h aq0 = load_a_frag(qh + (size_t)qs0 * T5_D, T5_D, 0, 0);
  v16h aq1 = load_a_frag(qh + (size_t)qs0 * T5_D, T5_D, 0, 32);

  // Prologue: TDM tile 0 into buffer 0 (one wave issues; EXEC-independent).
  if (w == 0) {
    tdm_load_2d(kh,  &Ks[0][0], T5_D, T5_S, 64, 64, T5_D);  // K: rows t
    tdm_load_2d(vth, &Vs[0][0], T5_S, T5_D, 64, 64, T5_S);  // V^T: rows d
  }

  float mstat[8], lstat[8];
  v8f o[4] = {};
#pragma unroll
  for (int r = 0; r < 8; ++r) { mstat[r] = -1e30f; lstat[r] = 0.f; }

  for (int t0 = 0; t0 < T5_S; t0 += 64) {
    const int buf = (t0 >> 6) & 1;
    __syncthreads();  // everyone done reading buf^1 from iteration t0-64
    if (w == 0) {
      if (t0 + 64 < T5_S) {
        // Prefetch next tile into the other buffer, then wait for the
        // current tile (in-order completion: <=2 outstanding => current done).
        tdm_load_2d(kh + (size_t)(t0 + 64) * T5_D, &Ks[buf ^ 1][0],
                    T5_D, T5_S, 64, 64, T5_D);
        tdm_load_2d(vth + (t0 + 64), &Vs[buf ^ 1][0],
                    T5_S, T5_D, 64, 64, T5_S);
        asm volatile("s_wait_tensorcnt 0x2" ::: "memory");
      } else {
        asm volatile("s_wait_tensorcnt 0x0" ::: "memory");
      }
    }
    __syncthreads();  // publish current tile to all waves

    // Prefetch next tile's position-bias rows (dominant HBM stream, 256 MB).
    if (t0 + 64 < T5_S) {
#pragma unroll
      for (int r = 0; r < 8; ++r)
        __builtin_prefetch(
            &pbh[(size_t)(qs0 + r + lhi) * T5_S + t0 + 64 + ln * 4], 0, 3);
    }

    // Scores: 16x64 tile = 4 x (16x16), K fragments from LDS.
    v8f sc[4];
#pragma unroll
    for (int tj = 0; tj < 4; ++tj) {
      v8f s = {};
      s = wmma16(aq0, load_b_frag(&Ks[buf][0], 64, tj * 16, 0), s);
      s = wmma16(aq1, load_b_frag(&Ks[buf][0], 64, tj * 16, 32), s);
      sc[tj] = s;
    }
    // Position bias (T5: no 1/sqrt(d) scaling).
#pragma unroll
    for (int tj = 0; tj < 4; ++tj) {
      int tcol = t0 + tj * 16 + ln;
#pragma unroll
      for (int r = 0; r < 8; ++r)
        sc[tj][r] += pbh[(size_t)(qs0 + r + lhi) * T5_S + tcol];
    }

    // Online softmax; row M lives in one 16-lane half -> xor masks 8..1.
#pragma unroll
    for (int r = 0; r < 8; ++r) {
      float m = fmaxf(fmaxf(sc[0][r], sc[1][r]), fmaxf(sc[2][r], sc[3][r]));
#pragma unroll
      for (int mask = 8; mask >= 1; mask >>= 1)
        m = fmaxf(m, __shfl_xor(m, mask, 32));
      float mnew  = fmaxf(mstat[r], m);
      float scale = __expf(mstat[r] - mnew);
      mstat[r] = mnew;
      float sum = 0.f;
#pragma unroll
      for (int tj = 0; tj < 4; ++tj) {
        float p = __expf(sc[tj][r] - mnew);
        sc[tj][r] = p;
        sum += p;
      }
#pragma unroll
      for (int mask = 8; mask >= 1; mask >>= 1)
        sum += __shfl_xor(sum, mask, 32);
      lstat[r] = lstat[r] * scale + sum;
#pragma unroll
      for (int dj = 0; dj < 4; ++dj) o[dj][r] *= scale;
    }

    // C-layout P -> wave-private LDS -> A-layout fragments.
    f16* pw = &Pbuf[w][0];
#pragma unroll
    for (int tj = 0; tj < 4; ++tj)
#pragma unroll
      for (int r = 0; r < 8; ++r)
        pw[(r + lhi) * 64 + tj * 16 + ln] = (f16)sc[tj][r];

    v16h ap[4];
#pragma unroll
    for (int tj = 0; tj < 2; ++tj) ap[tj] = load_a_frag(pw, 64, 0, tj * 32);

    // O += P @ V ; B fragments from [d][t] V tile (contiguous along t).
#pragma unroll
    for (int dj = 0; dj < 4; ++dj)
#pragma unroll
      for (int tj = 0; tj < 2; ++tj)
        o[dj] = wmma16(ap[tj], load_b_frag(&Vs[buf][0], 64, dj * 16, tj * 32),
                       o[dj]);
  }

  // Epilogue: normalize, write fp32 [B,S,DM].
#pragma unroll
  for (int dj = 0; dj < 4; ++dj)
#pragma unroll
    for (int r = 0; r < 8; ++r) {
      int srow = qs0 + r + lhi;
      int d    = dj * 16 + ln;
      out[((size_t)b * T5_S + srow) * T5_DM + h * T5_D + d] =
          o[dj][r] / lstat[r];
    }
}

// ------------------------- launch ------------------------------------------

extern "C" void kernel_launch(void* const* d_in, const int* in_sizes, int n_in,
                              void* d_out, int out_size, void* d_ws,
                              size_t ws_size, hipStream_t stream) {
  const float* hidden = (const float*)d_in[0];
  const float* pbias  = (const float*)d_in[1];
  const float* Wq     = (const float*)d_in[2];
  const float* Wk     = (const float*)d_in[3];
  const float* Wv     = (const float*)d_in[4];
  float* out = (float*)d_out;

  const size_t elems = (size_t)T5_B * T5_S * T5_DM;  // 8,388,608
  f16* q = (f16*)d_ws;
  f16* k = q + elems;
  f16* v = k + elems;  // stored transposed [B,H,D,S]

  dim3 pgrid(T5_DM / 64, (T5_B * T5_S) / 128, 3);
  t5_proj_kernel<<<pgrid, 256, 0, stream>>>(hidden, Wq, Wk, Wv, q, k, v);

  dim3 agrid(T5_S / 64, T5_B * T5_H);
  t5_attn_kernel<<<agrid, 128, 0, stream>>>(q, k, v, pbias, out);
}
